// MultiHeadAttention_37168646980166
// MI455X (gfx1250) — matrix operand order repaired and checked by reading, not compiled, not run on hardware
//
#include <hip/hip_runtime.h>

typedef unsigned short u16;
typedef __attribute__((ext_vector_type(16))) __bf16 bf16x16;
typedef __attribute__((ext_vector_type(8)))  float  floatx8;

#define TM 128
#define TN 128
#define TK 32

__device__ __forceinline__ u16 f32_to_bf16(float f) {
  union { float f; unsigned int u; } c; c.f = f;
  unsigned int u = c.u;
  u += 0x7fffu + ((u >> 16) & 1u);   // round-to-nearest-even
  return (u16)(u >> 16);
}

struct Frag { uint4 lo, hi; };       // 32 bytes == bf16x16

// ---------------------------------------------------------------------------
// Elementwise f32 -> bf16 cast
// ---------------------------------------------------------------------------
__global__ __launch_bounds__(256)
void cast_f32_bf16(const float* __restrict__ src, u16* __restrict__ dst, long n) {
  long i = (long)blockIdx.x * 256 + threadIdx.x;
  long stride = (long)gridDim.x * 256;
  for (; i < n; i += stride) dst[i] = f32_to_bf16(src[i]);
}

// ---------------------------------------------------------------------------
// Row softmax: f32 scores in, bf16 probabilities out. One block per row.
// ---------------------------------------------------------------------------
__global__ __launch_bounds__(256)
void softmax_bf16(const float* __restrict__ S, u16* __restrict__ P, int cols, float scale) {
  __shared__ float red[256];
  const size_t row = blockIdx.x;
  const float* s = S + row * (size_t)cols;
  u16* p = P + row * (size_t)cols;
  const int tid = threadIdx.x;

  float vals[8];            // cols <= 2048
  int cnt = 0;
  float lmax = -3.4e38f;
  for (int c = tid; c < cols; c += 256) {
    float x = s[c] * scale;
    vals[cnt++] = x;
    lmax = fmaxf(lmax, x);
  }
  red[tid] = lmax; __syncthreads();
  for (int o = 128; o > 0; o >>= 1) {
    if (tid < o) red[tid] = fmaxf(red[tid], red[tid + o]);
    __syncthreads();
  }
  float rmax = red[0]; __syncthreads();

  float lsum = 0.f;
  for (int i = 0; i < cnt; ++i) { vals[i] = __expf(vals[i] - rmax); lsum += vals[i]; }
  red[tid] = lsum; __syncthreads();
  for (int o = 128; o > 0; o >>= 1) {
    if (tid < o) red[tid] += red[tid + o];
    __syncthreads();
  }
  float inv = 1.0f / red[0];
  cnt = 0;
  for (int c = tid; c < cols; c += 256) p[c] = f32_to_bf16(vals[cnt++] * inv);
}

// ---------------------------------------------------------------------------
// Generic batched bf16 GEMM with f32 accumulation via V_WMMA_F32_16X16X32_BF16.
// C[g] = A[g](MxK) * B[g](KxN)   (B optionally stored transposed: NxK)
// Per-batch operand base = ptr + ((g / div) % mod) * stride.
// Block: 256 threads (8 wave32), tile 128x128, K-step 32, LDS double-buffered.
// Tile staging uses GLOBAL_LOAD_ASYNC_TO_LDS_B128 (ASYNCcnt) where the global
// layout permits a straight copy; the K-major B case scatter-transposes by hand.
// Requires: K % 32 == 0, M % 16 == 0, Nd % 16 == 0 (guards handle partial tiles).
// ---------------------------------------------------------------------------
__global__ __launch_bounds__(256, 2)
void gemm_bf16_wmma(const u16* __restrict__ Abase, long a_stride, int a_div, int a_mod,
                    const u16* __restrict__ Bbase, long b_stride, int b_div, int b_mod,
                    void* __restrict__ Cbase, long c_stride, int c_div, int c_mod,
                    int M, int Nd, int K, int lda, int ldb, int ldc,
                    int transB, int c_bf16) {
  __shared__ __align__(16) u16 lsA[2][TM][TK];        // A tile, row(M)-major
  __shared__ __align__(16) u16 lsB[2][TN][TK];        // B tile, stored N-major (B^T)

  const int g = blockIdx.z;
  const u16* A  = Abase + (size_t)((g / a_div) % a_mod) * a_stride;
  const u16* Bm = Bbase + (size_t)((g / b_div) % b_mod) * b_stride;
  const size_t coff = (size_t)((g / c_div) % c_mod) * c_stride;

  const int m0   = blockIdx.y * TM;
  const int n0   = blockIdx.x * TN;
  const int tid  = threadIdx.x;
  const int lane = tid & 31;
  const int wave = tid >> 5;
  const int wm   = wave & 3;    // 4 wave-rows of 32
  const int wn   = wave >> 2;   // 2 wave-cols of 64

  auto load_tiles = [&](int buf, int k0) {
    // ---- A tile: 32B/thread, async copy global -> LDS (no VGPR staging) ----
    {
      int row  = tid >> 1;
      int half = tid & 1;
      unsigned lp = (unsigned)(uintptr_t)&lsA[buf][row][half * 16];
      if (m0 + row < M) {
        const u16* gp = A + (size_t)(m0 + row) * lda + k0 + half * 16;
        asm volatile(
            "global_load_async_to_lds_b128 %0, %1, off\n\t"
            "global_load_async_to_lds_b128 %0, %1, off offset:16"
            :: "v"(lp), "v"(gp) : "memory");
        if (k0 + TK < K) __builtin_prefetch(gp + TK, 0, 1);   // next K-tile
      } else {
        uint4 z = make_uint4(0u, 0u, 0u, 0u);
        uint4* d = reinterpret_cast<uint4*>(&lsA[buf][row][half * 16]);
        d[0] = z; d[1] = z;
      }
    }
    // ---- B tile into N-major LDS ----
    if (transB) {                 // B stored NxK: rows contiguous in K, async copy
      int row  = tid >> 1;
      int half = tid & 1;
      unsigned lp = (unsigned)(uintptr_t)&lsB[buf][row][half * 16];
      if (n0 + row < Nd) {
        const u16* gp = Bm + (size_t)(n0 + row) * ldb + k0 + half * 16;
        asm volatile(
            "global_load_async_to_lds_b128 %0, %1, off\n\t"
            "global_load_async_to_lds_b128 %0, %1, off offset:16"
            :: "v"(lp), "v"(gp) : "memory");
        if (k0 + TK < K) __builtin_prefetch(gp + TK, 0, 1);
      } else {
        uint4 z = make_uint4(0u, 0u, 0u, 0u);
        uint4* d = reinterpret_cast<uint4*>(&lsB[buf][row][half * 16]);
        d[0] = z; d[1] = z;
      }
    } else {                      // B stored KxN: read 16 along N, scatter-transpose
      int kk = tid >> 3;          // 0..31
      int n  = (tid & 7) * 16;    // 0..112
      uint4 v0 = make_uint4(0u, 0u, 0u, 0u), v1 = make_uint4(0u, 0u, 0u, 0u);
      if (n0 + n < Nd) {          // Nd is a multiple of 16 -> chunk fully in/out
        const uint4* p = reinterpret_cast<const uint4*>(Bm + (size_t)(k0 + kk) * ldb + n0 + n);
        v0 = p[0]; v1 = p[1];
        if (k0 + TK < K) __builtin_prefetch(Bm + (size_t)(k0 + TK + kk) * ldb + n0 + n, 0, 1);
      }
      __align__(16) u16 tmp[16];
      *reinterpret_cast<uint4*>(&tmp[0]) = v0;
      *reinterpret_cast<uint4*>(&tmp[8]) = v1;
      #pragma unroll
      for (int j = 0; j < 16; ++j) lsB[buf][n + j][kk] = tmp[j];
    }
  };

  floatx8 zero;
  #pragma unroll
  for (int e = 0; e < 8; ++e) zero[e] = 0.0f;
  floatx8 acc[2][4];
  #pragma unroll
  for (int i = 0; i < 2; ++i)
    #pragma unroll
    for (int j = 0; j < 4; ++j) acc[i][j] = zero;

  const int kTiles = K / TK;
  load_tiles(0, 0);
  asm volatile("s_wait_asynccnt 0x0" ::: "memory");   // LDS data landed
  __syncthreads();

  for (int kt = 0; kt < kTiles; ++kt) {
    const int cur = kt & 1;
    if (kt + 1 < kTiles) load_tiles(cur ^ 1, (kt + 1) * TK);

    // --- build fragments from LDS in contiguous K order (same for A and B) ---
    // A (16x32): lane<16 holds K 0..15, lane>=16 holds K 16..31
    const int g0   = (lane < 16) ? 0 : 16;
    const int mrow = wm * 32 + (lane & 15);
    bf16x16 av[2];
    #pragma unroll
    for (int ms = 0; ms < 2; ++ms) {
      const uint4* p = reinterpret_cast<const uint4*>(&lsA[cur][mrow + ms * 16][g0]);
      Frag f; f.lo = p[0]; f.hi = p[1];          // K g0..g0+15
      av[ms] = __builtin_bit_cast(bf16x16, f);
    }
    // B (32x16, N-major LDS): lane<16 -> K 0..15, lane>=16 -> K 16..31, N = lane&15
    const int klo = (lane < 16) ? 0 : 16;
    bf16x16 bv[4];
    #pragma unroll
    for (int ns = 0; ns < 4; ++ns) {
      const uint4* p = reinterpret_cast<const uint4*>(&lsB[cur][wn * 64 + ns * 16 + (lane & 15)][klo]);
      Frag f; f.lo = p[0]; f.hi = p[1];
      bv[ns] = __builtin_bit_cast(bf16x16, f);
    }

    #pragma unroll
    for (int ms = 0; ms < 2; ++ms)
      #pragma unroll
      for (int ns = 0; ns < 4; ++ns)
        acc[ms][ns] = __builtin_amdgcn_wmma_f32_16x16x32_bf16(
            false, av[ms], false, bv[ns], (short)0, acc[ms][ns], false, false);

    asm volatile("s_wait_asynccnt 0x0" ::: "memory"); // next tile fully in LDS
    __syncthreads();
  }

  // --- epilogue: C/D layout -> VGPR r holds M = r (+8 for lanes 16..31) ---
  const int rbase = (lane < 16) ? 0 : 8;
  const int ccol  = lane & 15;
  if (c_bf16) {
    u16* C = (u16*)Cbase + coff;
    #pragma unroll
    for (int ms = 0; ms < 2; ++ms)
      #pragma unroll
      for (int ns = 0; ns < 4; ++ns) {
        const int col = n0 + wn * 64 + ns * 16 + ccol;
        if (col < Nd) {
          const int row0 = m0 + wm * 32 + ms * 16 + rbase;
          #pragma unroll
          for (int r = 0; r < 8; ++r) {
            const int row = row0 + r;
            if (row < M) C[(size_t)row * ldc + col] = f32_to_bf16(acc[ms][ns][r]);
          }
        }
      }
  } else {
    float* C = (float*)Cbase + coff;
    #pragma unroll
    for (int ms = 0; ms < 2; ++ms)
      #pragma unroll
      for (int ns = 0; ns < 4; ++ns) {
        const int col = n0 + wn * 64 + ns * 16 + ccol;
        if (col < Nd) {
          const int row0 = m0 + wm * 32 + ms * 16 + rbase;
          #pragma unroll
          for (int r = 0; r < 8; ++r) {
            const int row = row0 + r;
            if (row < M) C[(size_t)row * ldc + col] = acc[ms][ns][r];
          }
        }
      }
  }
}

// ---------------------------------------------------------------------------
// Host launch
// ---------------------------------------------------------------------------
static inline size_t align256(size_t x) { return (x + 255) & ~(size_t)255; }

static void launch_gemm(hipStream_t s,
                        const u16* A, long a_stride, int a_div, int a_mod,
                        const u16* Bm, long b_stride, int b_div, int b_mod,
                        void* C, long c_stride, int c_div, int c_mod,
                        int M, int Nd, int K, int lda, int ldb, int ldc,
                        int transB, int c_bf16, int G) {
  dim3 grid((Nd + TN - 1) / TN, (M + TM - 1) / TM, G);
  gemm_bf16_wmma<<<grid, 256, 0, s>>>(A, a_stride, a_div, a_mod,
                                      Bm, b_stride, b_div, b_mod,
                                      C, c_stride, c_div, c_mod,
                                      M, Nd, K, lda, ldb, ldc, transB, c_bf16);
}

extern "C" void kernel_launch(void* const* d_in, const int* in_sizes, int n_in,
                              void* d_out, int out_size, void* d_ws, size_t ws_size,
                              hipStream_t stream) {
  (void)in_sizes; (void)n_in; (void)out_size; (void)ws_size;
  constexpr int Bz = 4, Nn = 1024, DM = 1024, Hh = 16, DK = 64, DV = 1024;
  constexpr int G = Bz * Hh;

  const float* q  = (const float*)d_in[0];
  const float* k  = (const float*)d_in[1];
  const float* v  = (const float*)d_in[2];
  const float* wq = (const float*)d_in[3];
  const float* wk = (const float*)d_in[4];
  const float* wv = (const float*)d_in[5];
  const float* wo = (const float*)d_in[6];

  char* ws = (char*)d_ws;
  size_t off = 0;
  auto take = [&](size_t elems, size_t esize) -> void* {
    void* p = ws + off;
    off = align256(off + elems * esize);
    return p;
  };

  u16*   xq  = (u16*)take((size_t)Bz * Nn * DM, 2);
  u16*   xk  = (u16*)take((size_t)Bz * Nn * DM, 2);
  u16*   xv  = (u16*)take((size_t)Bz * Nn * DM, 2);
  u16*   wqb = (u16*)take((size_t)Hh * DM * DK, 2);
  u16*   wkb = (u16*)take((size_t)Hh * DM * DK, 2);
  u16*   wvb = (u16*)take((size_t)Hh * DM * DV, 2);
  u16*   wob = (u16*)take((size_t)DM * Hh * DV, 2);
  u16*   qp  = (u16*)take((size_t)G * Nn * DK, 2);
  u16*   kp  = (u16*)take((size_t)G * Nn * DK, 2);
  u16*   vp  = (u16*)take((size_t)G * Nn * DV, 2);
  float* Sf  = (float*)take((size_t)G * Nn * Nn, 4);
  u16*   Pw  = (u16*)take((size_t)G * Nn * Nn, 2);
  u16*   o1  = (u16*)take((size_t)G * Nn * DV, 2);

  // f32 -> bf16 casts
  cast_f32_bf16<<<4096, 256, 0, stream>>>(q,  xq,  (long)Bz * Nn * DM);
  cast_f32_bf16<<<4096, 256, 0, stream>>>(k,  xk,  (long)Bz * Nn * DM);
  cast_f32_bf16<<<4096, 256, 0, stream>>>(v,  xv,  (long)Bz * Nn * DM);
  cast_f32_bf16<<<4096, 256, 0, stream>>>(wq, wqb, (long)Hh * DM * DK);
  cast_f32_bf16<<<4096, 256, 0, stream>>>(wk, wkb, (long)Hh * DM * DK);
  cast_f32_bf16<<<4096, 256, 0, stream>>>(wv, wvb, (long)Hh * DM * DV);
  cast_f32_bf16<<<4096, 256, 0, stream>>>(wo, wob, (long)DM * Hh * DV);

  // Q = query[b] (NxDM) x w_q[h] (DMxDK) ; batch g = b*H + h
  launch_gemm(stream, xq, (long)Nn * DM, Hh, Bz, wqb, (long)DM * DK, 1, Hh,
              qp, (long)Nn * DK, 1, G, Nn, DK, DM, DM, DK, DK, 0, 1, G);
  // K projection
  launch_gemm(stream, xk, (long)Nn * DM, Hh, Bz, wkb, (long)DM * DK, 1, Hh,
              kp, (long)Nn * DK, 1, G, Nn, DK, DM, DM, DK, DK, 0, 1, G);
  // V projection
  launch_gemm(stream, xv, (long)Nn * DM, Hh, Bz, wvb, (long)DM * DV, 1, Hh,
              vp, (long)Nn * DV, 1, G, Nn, DV, DM, DM, DV, DV, 0, 1, G);
  // scores = Q x K^T  (f32 out)
  launch_gemm(stream, qp, (long)Nn * DK, 1, G, kp, (long)Nn * DK, 1, G,
              Sf, (long)Nn * Nn, 1, G, Nn, Nn, DK, DK, DK, Nn, 1, 0, G);
  // softmax rows, scale = 1/sqrt(DK)
  softmax_bf16<<<G * Nn, 256, 0, stream>>>(Sf, Pw, Nn, 0.125f);
  // attn_out = P x V   (bf16 out)
  launch_gemm(stream, Pw, (long)Nn * Nn, 1, G, vp, (long)Nn * DV, 1, G,
              o1, (long)Nn * DV, 1, G, Nn, DV, Nn, Nn, DV, DV, 0, 1, G);
  // final = w_o (DM x H*DV) x out[b] (H*N x DV) -> d_out [B, DM, DV] f32
  launch_gemm(stream, wob, 0, 1, 1, o1, (long)Hh * Nn * DV, 1, Bz,
              d_out, (long)DM * DV, 1, Bz, DM, DV, Hh * Nn,
              Hh * DV, DV, DV, 0, 0, Bz);
}
